// ProbabilisticAutoma_62380105007700
// MI455X (gfx1250) — matrix operand order repaired and checked by reading, not compiled, not run on hardware
//
#include <hip/hip_runtime.h>
#include <hip/hip_bf16.h>
#include <stdint.h>

// ProbabilisticAutoma on MI455X (gfx1250, wave32, WMMA).
// B=256 chains, T=2048 serial steps, S=64 states, A=16 actions, R=2.
//
// v2 changes (from asm feedback):
//  - A operand loaded from LDS ONCE per step (4 ds_load_b128), per-action row
//    masking is now register v_and_b32 (per-lane-uniform mask) -> removes the
//    serialized ds_load + s_wait_dscnt round trip per present action.
//  - Action indices software-pipelined: t+1's load issues at the top of step t
//    and is consumed next step, hiding global/L2 latency behind the step.

typedef __attribute__((ext_vector_type(16))) _Float16 v16h;
typedef __attribute__((ext_vector_type(8)))  float    v8f;

#define B_  256
#define T_  2048
#define A_  16
#define S_  64
#define R_  2

union ABu { uint4 q[2]; v16h h; };

__device__ __forceinline__ uint4 and4(uint4 a, unsigned m) {
  return make_uint4(a.x & m, a.y & m, a.z & m, a.w & m);
}

// wave32 xor-swizzle reduction helper (group-of-32 ds_swizzle: and=0x1f)
template<int XOR>
__device__ __forceinline__ float swzxor(float x) {
  int i = __builtin_amdgcn_ds_swizzle(__builtin_bit_cast(int, x),
                                      (XOR << 10) | 0x1f);
  return __builtin_bit_cast(float, i);
}

// ---------------------------------------------------------------------------
// Prepass: one-hot [B,T,16] f32 -> action index, stored [T,B] so the scan's
// per-step fetch for a 16-chain block is one contiguous 64B line.
// ---------------------------------------------------------------------------
__global__ __launch_bounds__(256) void onehot_to_idx(
    const float* __restrict__ oh, int* __restrict__ idx)
{
  int g = blockIdx.x * 256 + threadIdx.x;
  if (g >= B_ * T_) return;
  int b = g % B_;            // consecutive threads -> coalesced idx store
  int t = g / B_;
  const float* p = oh + ((size_t)b * T_ + t) * A_;  // 64B contiguous per thread
  int best = 0; float bv = p[0];
  #pragma unroll
  for (int i = 1; i < A_; ++i) { float v = p[i]; if (v > bv) { bv = v; best = i; } }
  idx[t * B_ + b] = best;
}

// ---------------------------------------------------------------------------
// Main scan kernel: grid = 16 blocks (16 chains each), 128 threads = 4 waves.
// Wave w computes output columns [16w, 16w+16).
// ---------------------------------------------------------------------------
__global__ __launch_bounds__(128) void automaton_scan(
    const float* __restrict__ trans,    // [A,S,S] f32
    const float* __restrict__ rewm,     // [S,R]   f32
    const int*   __restrict__ idxbuf,   // [T,B]   int
    float* __restrict__ statesOut,      // [B,T,S] f32
    float* __restrict__ rewsOut)        // [B,T,R] f32
{
  // stateA: f16 state, row-major [row][k]; A-operand loads are ds_load_b128.
  __shared__ __align__(16) _Float16 stateA[16 * S_];
  __shared__ __align__(16) float    nsbuf[16 * S_];   // pre-softmax ns[16][64]

  const int tid   = threadIdx.x;
  const int lane  = tid & 31;
  const int ntile = tid >> 5;             // wave id = N-tile 0..3
  const int blk   = blockIdx.x;           // chain block (16 chains)
  const int hi    = (lane >> 4) & 1;      // lane group (ISA A/B/C layouts)
  const int zc    = ntile * 16 + (lane & 15);   // this lane's output column

  // --- Resident B operands: f16 slice of every P[a] for this wave's N-tile.
  // B layout (16-bit, 32x16/chunk): lane = N (l&15); VGPR0-3 = K kb..kb+7,
  // VGPR4-7 = K 16+kb.., kb = 8*hi (matches A).
  v16h Breg[A_][2];
  #pragma unroll
  for (int a = 0; a < A_; ++a) {
    #pragma unroll
    for (int kc = 0; kc < 2; ++kc) {
      v16h h;
      #pragma unroll
      for (int v = 0; v < 8; ++v) {
        #pragma unroll
        for (int e = 0; e < 2; ++e) {
          int s = kc * 32 + hi * 8 + v * 2 + e + (v >= 4 ? 8 : 0);
          h[v * 2 + e] = (_Float16)trans[((a * S_) + s) * S_ + zc];
        }
      }
      Breg[a][kc] = h;
    }
  }

  // --- Softmax-phase geometry: 8 threads per row, 8 columns per thread.
  const int srow = tid >> 3;              // row 0..15
  const int c0   = (tid & 7) * 8;         // first of my 8 columns
  float rw0[8], rw1[8];
  #pragma unroll
  for (int i = 0; i < 8; ++i) {
    rw0[i] = rewm[(c0 + i) * R_ + 0];
    rw1[i] = rewm[(c0 + i) * R_ + 1];
  }

  // --- init: s0 = one-hot(state 0).
  for (int i = tid; i < 16 * S_; i += 128) stateA[i] = (_Float16)0.f;
  __syncthreads();
  if (tid < 16) stateA[tid * S_ + 0] = (_Float16)1.f;
  __syncthreads();

  const uint4* rp = (const uint4*)stateA + (lane & 15) * 8;  // my A row (128B)
  const int    actAddrBase = blk * 16 + (lane & 15);

  // software pipeline: action indices for step t loaded during step t-1
  int act_l = idxbuf[0 * B_ + actAddrBase];

  for (int t = 0; t < T_; ++t) {
    const int act_cur = act_l;
    if (t + 1 < T_) act_l = idxbuf[(t + 1) * B_ + actAddrBase];  // hidden load

    // A operand (this lane's state row), loaded once per step:
    // chunk0 = K 0..31, chunk1 = K 32..63 (A layout: VGPR0-3 = K kb..kb+7,
    // VGPR4-7 = K 16+kb.., kb = 8*hi)
    const uint4 q0 = rp[hi];     const uint4 q1 = rp[hi + 2];
    const uint4 q2 = rp[hi + 4]; const uint4 q3 = rp[hi + 6];

    v8f acc = {0.f, 0.f, 0.f, 0.f, 0.f, 0.f, 0.f, 0.f};

    #pragma unroll
    for (int a = 0; a < A_; ++a) {
      // uniform skip of actions no chain in this block took this step
      if (__builtin_amdgcn_ballot_w32(act_cur == a) != 0u) {
        // per-lane-uniform row mask (lane l owns state row l&15): pure VALU,
        // co-executes under the XDL WMMA pipe.
        const unsigned msk = (act_cur == a) ? 0xffffffffu : 0u;
        ABu m0, m1;
        m0.q[0] = and4(q0, msk); m0.q[1] = and4(q1, msk);
        m1.q[0] = and4(q2, msk); m1.q[1] = and4(q3, msk);
        acc = __builtin_amdgcn_wmma_f32_16x16x32_f16(
            false, m0.h, false, Breg[a][0], (short)0, acc, false, false);
        acc = __builtin_amdgcn_wmma_f32_16x16x32_f16(
            false, m1.h, false, Breg[a][1], (short)0, acc, false, false);
      }
    }

    // scatter C tile (VGPR j, lane l) -> ns[row = j + 8*hi][zc]
    {
      const int rb = hi * 8;
      #pragma unroll
      for (int j = 0; j < 8; ++j) nsbuf[(rb + j) * S_ + zc] = acc[j];
    }
    __syncthreads();

    // ---- softmax + reward + emit: 8 lanes/row, xor-reductions over 1,2,4
    float4 x0 = *(const float4*)&nsbuf[srow * S_ + c0];
    float4 x1 = *(const float4*)&nsbuf[srow * S_ + c0 + 4];
    float xs[8] = {x0.x, x0.y, x0.z, x0.w, x1.x, x1.y, x1.z, x1.w};

    float m = xs[0];
    #pragma unroll
    for (int i = 1; i < 8; ++i) m = fmaxf(m, xs[i]);
    m = fmaxf(m, swzxor<1>(m)); m = fmaxf(m, swzxor<2>(m)); m = fmaxf(m, swzxor<4>(m));

    float r0 = 0.f, r1 = 0.f, sum = 0.f, e[8];
    #pragma unroll
    for (int i = 0; i < 8; ++i) {
      r0 += xs[i] * rw0[i];            // reward uses PRE-softmax ns
      r1 += xs[i] * rw1[i];
      e[i] = __expf(xs[i] - m);
      sum += e[i];
    }
    sum += swzxor<1>(sum); sum += swzxor<2>(sum); sum += swzxor<4>(sum);
    r0  += swzxor<1>(r0);  r0  += swzxor<2>(r0);  r0  += swzxor<4>(r0);
    r1  += swzxor<1>(r1);  r1  += swzxor<2>(r1);  r1  += swzxor<4>(r1);

    const float inv = 1.0f / sum;
    float p[8];
    #pragma unroll
    for (int i = 0; i < 8; ++i) p[i] = e[i] * inv;

    const size_t bt = (size_t)(blk * 16 + srow) * T_ + t;
    *(float4*)&statesOut[bt * S_ + c0]     = make_float4(p[0], p[1], p[2], p[3]);
    *(float4*)&statesOut[bt * S_ + c0 + 4] = make_float4(p[4], p[5], p[6], p[7]);
    if ((tid & 7) == 0) {
      rewsOut[bt * R_ + 0] = r0;
      rewsOut[bt * R_ + 1] = r1;
    }

    // next-step f16 A state back into LDS (8 contiguous halfs = one b128)
    union { uint4 q; _Float16 hh[8]; } pk;
    #pragma unroll
    for (int i = 0; i < 8; ++i) pk.hh[i] = (_Float16)p[i];
    *(uint4*)&stateA[srow * S_ + c0] = pk.q;
    __syncthreads();
  }
}

// ---------------------------------------------------------------------------
extern "C" void kernel_launch(void* const* d_in, const int* in_sizes, int n_in,
                              void* d_out, int out_size, void* d_ws, size_t ws_size,
                              hipStream_t stream) {
  const float* action_seq = (const float*)d_in[0];   // [B,T,A] f32 one-hot
  const float* trans_prob = (const float*)d_in[1];   // [A,S,S] f32
  const float* rew_matrix = (const float*)d_in[2];   // [S,R]   f32
  // d_in[3] = temp (unused, matching reference)

  int*   idxbuf    = (int*)d_ws;                     // [T,B] = 2 MB scratch
  float* statesOut = (float*)d_out;                  // [B,T,S]
  float* rewsOut   = statesOut + (size_t)B_ * T_ * S_;  // [B,T,R]

  onehot_to_idx<<<dim3((B_ * T_) / 256), dim3(256), 0, stream>>>(
      action_seq, idxbuf);
  automaton_scan<<<dim3(B_ / 16), dim3(128), 0, stream>>>(
      trans_prob, rew_matrix, idxbuf, statesOut, rewsOut);
}
